// ScannedMultiLayerLSTM_65438121722404
// MI455X (gfx1250) — compile-verified, matrix operand order checked by reading, not run on hardware
//
#include <hip/hip_runtime.h>
#include <hip/hip_bf16.h>
#include <stddef.h>

// Problem constants (from reference): T=64, B=32, W=1024, L=4, G=4*W=4096
#define TT 64
#define BB 32
#define WW 1024
#define LL 4
#define GG 4096

typedef __attribute__((ext_vector_type(16))) __bf16 v16bf;
typedef __attribute__((ext_vector_type(8)))  __bf16 v8bf;
typedef __attribute__((ext_vector_type(8)))  float  v8f;
typedef __attribute__((ext_vector_type(4)))  int    v4i;

#if __has_builtin(__builtin_amdgcn_global_load_async_to_lds_b128) && \
    __has_builtin(__builtin_amdgcn_s_wait_asynccnt)
#define USE_ASYNC_LDS 1
typedef __attribute__((address_space(1))) v4i g_v4i;   // global int4
typedef __attribute__((address_space(3))) v4i l_v4i;   // LDS int4
#endif

// ---------------------------------------------------------------------------
// Prologue 1: tiled transpose + f32 -> bf16 convert of weights.
// in  : [L][K=WW (k)][N=GG (n)] f32 (n contiguous)
// out : [L][N=GG (n)][K=WW (k)] bf16 (k contiguous)
// ---------------------------------------------------------------------------
__global__ void wt_transpose_bf16(const float* __restrict__ in,
                                  __bf16* __restrict__ out) {
    __shared__ __bf16 tile[32][33];
    const int l  = blockIdx.z;
    const int n0 = blockIdx.x * 32;
    const int k0 = blockIdx.y * 32;
    const int tx = threadIdx.x & 31;
    const int ty = threadIdx.x >> 5;            // 0..7
    const float*  in_l  = in  + (size_t)l * WW * GG;
    __bf16*       out_l = out + (size_t)l * GG * WW;
#pragma unroll
    for (int p = 0; p < 4; ++p) {
        const int r = ty + p * 8;
        tile[r][tx] = (__bf16)in_l[(size_t)(k0 + r) * GG + (n0 + tx)];
    }
    __syncthreads();
#pragma unroll
    for (int p = 0; p < 4; ++p) {
        const int r = ty + p * 8;
        out_l[(size_t)(n0 + r) * WW + (k0 + tx)] = tile[tx][r];
    }
}

// ---------------------------------------------------------------------------
// Prologue 2: f32 -> bf16 elementwise convert (used for x and h0).
// ---------------------------------------------------------------------------
__global__ void cvt_bf16(const float* __restrict__ in, __bf16* __restrict__ out,
                         int n) {
    const int i = blockIdx.x * blockDim.x + threadIdx.x;
    if (i < n) out[i] = (__bf16)in[i];
}

// Prologue 3: copy f32 c0 into ping-pong slot 1 (t=0 reads parity 1).
__global__ void copy_f32(const float* __restrict__ in, float* __restrict__ out,
                         int n) {
    const int i = blockIdx.x * blockDim.x + threadIdx.x;
    if (i < n) out[i] = in[i];
}

// ---------------------------------------------------------------------------
// Fragment builder: 16 bf16 per lane = two 128-bit loads (K..K+7, K+16..K+23
// for lanes 0-15; +8 for lanes 16-31), matching the 16-bit WMMA VGPR layout.
// ---------------------------------------------------------------------------
__device__ __forceinline__ v16bf frag16(const __bf16* p) {
    v8bf lo = *(const v8bf*)(p);
    v8bf hi = *(const v8bf*)(p + 16);
    return __builtin_shufflevector(lo, hi, 0,1,2,3,4,5,6,7,8,9,10,11,12,13,14,15);
}

__device__ __forceinline__ float sigm(float x) {
    return 1.0f / (1.0f + __expf(-x));
}
__device__ __forceinline__ float tanh_fast(float x) {
    return 2.0f / (1.0f + __expf(-2.0f * x)) - 1.0f;
}

// ---------------------------------------------------------------------------
// One (t, l) layer step.
//   gates(32 x 64-slice) = y(32x1024) @ WxT' + mask(h_prev) @ WhT' + b
// Block owns 16 hidden units -> 4 gates x 16 cols = 8 WMMA tiles; 8 waves ->
// one 16x16 tile per wave; K=1024 -> 64 v_wmma_f32_16x16x32_bf16 per wave.
// grid = 64 blocks, block = 256 threads (8 waves).
// ---------------------------------------------------------------------------
__global__ __launch_bounds__(256)
void lstm_layer_step(const __bf16* __restrict__ y_in,       // [32][1024] bf16
                     const __bf16* __restrict__ h_prev,     // [32][1024] bf16
                     const float* __restrict__ c_prev,      // [32][1024] f32
                     const unsigned char* __restrict__ rst, // [32] resets at t
                     const __bf16* __restrict__ wxt,        // [4096][1024] bf16
                     const __bf16* __restrict__ wht,        // [4096][1024] bf16
                     const float* __restrict__ bias,        // [4096]
                     float* __restrict__ c_cur,             // [32][1024] f32
                     __bf16* __restrict__ h_cur,            // [32][1024] bf16
                     float* __restrict__ y_out,             // ys[t] if top layer
                     float* __restrict__ c_fin,             // if t==T-1
                     float* __restrict__ h_fin) {
    __shared__ __bf16 yl[BB][WW];   // 64 KB
    __shared__ __bf16 hl[BB][WW];   // 64 KB
    __shared__ float  gl[BB][64];   //  8 KB gate tile: [row][gate*16 + jj]

    const int tid = threadIdx.x;
    __bf16* yflat = &yl[0][0];
    __bf16* hflat = &hl[0][0];

    // ---- Stage bf16 A operands to LDS (pure copy, 4096 x 16B chunks each) --
#if USE_ASYNC_LDS
#pragma unroll
    for (int i = 0; i < 16; ++i) {
        const int e = (tid + i * 256) * 8;     // bf16 element offset, 16B chunk
        __builtin_amdgcn_global_load_async_to_lds_b128(
            (g_v4i*)(y_in + e), (l_v4i*)(yflat + e), 0, 0);
        __builtin_amdgcn_global_load_async_to_lds_b128(
            (g_v4i*)(h_prev + e), (l_v4i*)(hflat + e), 0, 0);
    }
    __builtin_amdgcn_s_wait_asynccnt(0);
#else
#pragma unroll
    for (int i = 0; i < 16; ++i) {
        const int e = (tid + i * 256) * 8;
        *(v8bf*)(yflat + e) = *(const v8bf*)(y_in + e);
        *(v8bf*)(hflat + e) = *(const v8bf*)(h_prev + e);
    }
#endif
    __syncthreads();

    // ---- Apply reset mask: zero recurrent-h LDS rows for reset batches ----
    {
        const int row = tid >> 3;               // 32 rows x 8 threads
        const int seg = (tid & 7) * 128;        // 128 bf16 per thread
        if (rst[row] != 0) {
            const v8bf z = {};
#pragma unroll
            for (int q = 0; q < 16; ++q) *(v8bf*)(&hl[row][seg + q * 8]) = z;
        }
    }
    __syncthreads();

    // ---- WMMA: each wave computes one 16x16 gate tile ----
    const int lane = tid & 31;
    const int wave = tid >> 5;                  // 0..7
    const int g    = wave >> 1;                 // gate 0..3 (i,f,g,o)
    const int mt   = wave & 1;                  // batch-row tile 0..1
    const int lr   = lane & 15;
    const int half = lane >> 4;
    const int o1   = half * 8;                  // K sub-offset per lane group
    const int m    = mt * 16 + lr;              // A row (batch index)
    const size_t nrow = (size_t)(g * WW + blockIdx.x * 16 + lr); // gate col

    const __bf16* ya = &yl[m][o1];
    const __bf16* ha = &hl[m][o1];
    const __bf16* bx = wxt + nrow * WW + o1;
    const __bf16* bh = wht + nrow * WW + o1;

    v8f acc = {};
#pragma unroll 4
    for (int kk = 0; kk < WW; kk += 32) {
        v16bf a0 = frag16(ya + kk);
        v16bf b0 = frag16(bx + kk);
        acc = __builtin_amdgcn_wmma_f32_16x16x32_bf16(
                  false, a0, false, b0, (short)0, acc, false, false);
        v16bf a1 = frag16(ha + kk);
        v16bf b1 = frag16(bh + kk);
        acc = __builtin_amdgcn_wmma_f32_16x16x32_bf16(
                  false, a1, false, b1, (short)0, acc, false, false);
    }

    // Spill accumulator tile to LDS: VGPR r -> row (r or r+8), col = lr.
#pragma unroll
    for (int r = 0; r < 8; ++r) {
        gl[mt * 16 + half * 8 + r][g * 16 + lr] = acc[r];
    }
    __syncthreads();

    // ---- LSTM cell update: 32 rows x 16 hidden cols ----
    const int j0 = blockIdx.x * 16;
#pragma unroll
    for (int e = tid; e < BB * 16; e += 256) {
        const int row = e >> 4;
        const int jj  = e & 15;
        const int j   = j0 + jj;
        const float iv = gl[row][jj]       + bias[j];
        const float fv = gl[row][16 + jj]  + bias[WW + j];
        const float gv = gl[row][32 + jj]  + bias[2 * WW + j];
        const float ov = gl[row][48 + jj]  + bias[3 * WW + j];
        const float cp = (rst[row] != 0) ? 0.f : c_prev[(size_t)row * WW + j];
        const float cn = sigm(fv) * cp + sigm(iv) * tanh_fast(gv);
        const float hn = sigm(ov) * tanh_fast(cn);
        c_cur[(size_t)row * WW + j] = cn;
        h_cur[(size_t)row * WW + j] = (__bf16)hn;
        if (y_out) y_out[(size_t)row * WW + j] = hn;
        if (c_fin) { c_fin[(size_t)row * WW + j] = cn;
                     h_fin[(size_t)row * WW + j] = hn; }
    }
}

// ---------------------------------------------------------------------------
// Host-side launch. Inputs (setup_inputs order):
//  0: x (T,B,W) f32   1: resets (T,B) bool   2: c0 (L,B,W)   3: h0 (L,B,W)
//  4: Wx (L,W,4W)     5: Wh (L,W,4W)         6: b (L,4W)
// Output: ys (T,B,W) || cs_f (L,B,W) || hs_f (L,B,W), all f32.
// ---------------------------------------------------------------------------
extern "C" void kernel_launch(void* const* d_in, const int* in_sizes, int n_in,
                              void* d_out, int out_size, void* d_ws, size_t ws_size,
                              hipStream_t stream) {
    (void)in_sizes; (void)n_in; (void)out_size; (void)ws_size;

    const float*         x      = (const float*)d_in[0];
    const unsigned char* resets = (const unsigned char*)d_in[1];
    const float*         c0     = (const float*)d_in[2];
    const float*         h0     = (const float*)d_in[3];
    const float*         Wx     = (const float*)d_in[4];
    const float*         Wh     = (const float*)d_in[5];
    const float*         bias   = (const float*)d_in[6];
    float*               out    = (float*)d_out;

    const size_t WBYTES = (size_t)LL * GG * WW * sizeof(__bf16); // 33.5 MB each
    const size_t SB     = (size_t)BB * WW;                        // 32768 elems
    const size_t LSB    = (size_t)LL * SB;                        // 131072 elems

    char* ws = (char*)d_ws;
    __bf16* wxt   = (__bf16*)ws;                        ws += WBYTES;
    __bf16* wht   = (__bf16*)ws;                        ws += WBYTES;
    __bf16* x_bf  = (__bf16*)ws;                        ws += (size_t)TT * SB * 2;
    float*  cbuf[2]; cbuf[0] = (float*)ws;              ws += LSB * 4;
                     cbuf[1] = (float*)ws;              ws += LSB * 4;
    __bf16* hbuf[2]; hbuf[0] = (__bf16*)ws;             ws += LSB * 2;
                     hbuf[1] = (__bf16*)ws;             ws += LSB * 2;

    // Prologue: one-time weight transpose+convert, x convert, state init.
    dim3 tgrid(GG / 32, WW / 32, LL);
    wt_transpose_bf16<<<tgrid, 256, 0, stream>>>(Wx, wxt);
    wt_transpose_bf16<<<tgrid, 256, 0, stream>>>(Wh, wht);
    cvt_bf16<<<(int)((TT * SB + 255) / 256), 256, 0, stream>>>(
        x, x_bf, (int)(TT * SB));
    cvt_bf16<<<(int)((LSB + 255) / 256), 256, 0, stream>>>(
        h0, hbuf[1], (int)LSB);
    copy_f32<<<(int)((LSB + 255) / 256), 256, 0, stream>>>(
        c0, cbuf[1], (int)LSB);

    float* ys_base = out;                         // T*B*W
    float* cs_base = out + (size_t)TT * SB;       // L*B*W
    float* hs_base = cs_base + LSB;               // L*B*W

    for (int t = 0; t < TT; ++t) {
        const int p  = t & 1;
        const int pp = p ^ 1;
        for (int l = 0; l < LL; ++l) {
            const __bf16* yin = (l == 0) ? (x_bf + (size_t)t * SB)
                                         : (hbuf[p] + (size_t)(l - 1) * SB);
            float* yout = (l == LL - 1) ? (ys_base + (size_t)t * SB) : nullptr;
            float* cfin = (t == TT - 1) ? (cs_base + (size_t)l * SB) : nullptr;
            float* hfin = (t == TT - 1) ? (hs_base + (size_t)l * SB) : nullptr;
            lstm_layer_step<<<WW / 16, 256, 0, stream>>>(
                yin,
                hbuf[pp] + (size_t)l * SB,
                cbuf[pp] + (size_t)l * SB,
                resets + (size_t)t * BB,
                wxt + (size_t)l * GG * WW,
                wht + (size_t)l * GG * WW,
                bias + (size_t)l * GG,
                cbuf[p] + (size_t)l * SB,
                hbuf[p] + (size_t)l * SB,
                yout, cfin, hfin);
        }
    }
}
